// LinearChainCrf_36876589203579
// MI455X (gfx1250) — matrix operand order repaired and checked by reading, not compiled, not run on hardware
//
#include <hip/hip_runtime.h>
#include <hip/hip_bf16.h>

typedef __attribute__((ext_vector_type(16))) _Float16 v16h;
typedef __attribute__((ext_vector_type(8)))  float    v8f;

#define Bv 512
#define Tv 1024
#define Kv 64

#define LOG2E 1.4426950408889634f
#define LN2   0.6931471805599453f

// Raw hardware transcendentals: safe here because log arguments are
// guaranteed in [~0.8, ~80] (row-max stabilization => at least one p ~= 1,
// E = exp(trans) in [0.8, 1.25]) and exp arguments are <= 0 where
// underflow-to-zero is the desired semantics.
__device__ __forceinline__ float fast_exp(float x) {
    return __builtin_amdgcn_exp2f(x * LOG2E);
}
__device__ __forceinline__ float fast_log(float x) {
    return __builtin_amdgcn_logf(x) * LN2;
}

// ---------------------------------------------------------------------------
// Normalizer (forward algorithm) with WMMA.
// One workgroup = 16 batch rows, 4 waves; wave w owns output columns
// [16w, 16w+16) held permanently in a 16x16 f32 C/D fragment (8 VGPRs).
// Per step t:
//   stage1: partial row-max of own fragment (shfl_xor over lanes 0..15 halves)
//   stage2: combine row-max via LDS, p = exp(S - rm) -> f16 staged to LDS
//   stage3: load A fragments (16x32 f16) from LDS staging
//   stage4: acc = A0 @ Bf0 + A1 @ Bf1   (two v_wmma_f32_16x16x32_f16)
//   prefetch: em/mask for step t+1 (consumed next iteration)
//   stage5: S = rm + log(acc) + em[t]   (masked)
// ---------------------------------------------------------------------------
__global__ __launch_bounds__(128) void crf_forward_kernel(
    const float* __restrict__ em, const float* __restrict__ trans,
    const float* __restrict__ startT, const float* __restrict__ endT,
    const unsigned char* __restrict__ mask, float* __restrict__ part)
{
    __shared__ __align__(16) _Float16 Ef[64][64];   // exp(transitions), 8 KB
    __shared__ __align__(16) _Float16 ps[16][64];   // p staging, 2 KB
    __shared__ __align__(16) float    pmax[16][4];  // per-wave partial row max
    __shared__ __align__(16) float    psum[16][4];  // per-wave partial row sum

    const int tid  = threadIdx.x;
    const int wave = tid >> 5;        // 0..3 : column tile
    const int lane = tid & 31;
    const int n    = lane & 15;       // column within 16-wide tile
    const int hi   = lane >> 4;       // half-wave
    const int b0   = blockIdx.x * 16; // batch tile base

    // Build E = exp(transitions) in f16 (cooperative).
    for (int i = tid; i < 64 * 64; i += 128)
        Ef[i >> 6][i & 63] = (_Float16)fast_exp(trans[i]);
    __syncthreads();

    union H16 { v16h v; _Float16 h[16]; uint4 q[2]; };

    const int koff = hi * 8;          // A-matrix k sub-offset for this half
    const int jcol = wave * 16 + n;   // global output column

    // Register-resident B fragments: E[0:32, cols] and E[32:64, cols].
    // B 32x16 f16 layout (matches A): elem e<8 -> K=8*hi+e,
    // elem e>=8 -> K=16+8*hi+(e-8).
    H16 Bf0, Bf1;
    for (int e = 0; e < 16; ++e) {
        int k = (e < 8) ? (8 * hi + e) : (16 + 8 * hi + (e - 8));
        Bf0.h[e] = Ef[k][jcol];
        Bf1.h[e] = Ef[k + 32][jcol];
    }

    // Initial score: start[j] + emissions[b, 0, j], in C/D fragment layout.
    float S[8];
    const float startv = startT[jcol];
    for (int r = 0; r < 8; ++r) {
        int m = r + 8 * hi;
        S[r] = startv + em[((size_t)(b0 + m) * Tv) * Kv + jcol];
    }

    // Software-pipelined emission/mask prefetch (for step t = 1).
    float          evb[8];
    unsigned char  mkb[8];
    for (int r = 0; r < 8; ++r) {
        int m = r + 8 * hi;
        size_t row = (size_t)(b0 + m) * Tv + 1;
        evb[r] = em[row * Kv + jcol];
        mkb[r] = mask[row];
    }

    for (int t = 1; t < Tv; ++t) {
        // ---- stage1: partial row max over this wave's 16 columns ----
        float pm[8];
        for (int r = 0; r < 8; ++r) {
            float v = S[r];
            v = fmaxf(v, __shfl_xor(v, 1));
            v = fmaxf(v, __shfl_xor(v, 2));
            v = fmaxf(v, __shfl_xor(v, 4));
            v = fmaxf(v, __shfl_xor(v, 8));
            pm[r] = v;
        }
        if (n == 0)
            for (int r = 0; r < 8; ++r) pmax[r + 8 * hi][wave] = pm[r];
        __syncthreads();

        // ---- stage2: global row max, p = exp(S - rm) staged to LDS ----
        float rm[8];
        for (int r = 0; r < 8; ++r) {
            int m = r + 8 * hi;
            float4 q = *(const float4*)pmax[m];
            float v = fmaxf(fmaxf(q.x, q.y), fmaxf(q.z, q.w));
            rm[r] = v;
            ps[m][jcol] = (_Float16)fast_exp(S[r] - v);
        }
        __syncthreads();

        // ---- stage3: A fragments from staging (16-bit A 16x32 layout) ----
        H16 A0, A1;
        A0.q[0] = *(const uint4*)&ps[n][koff];           // K = koff..koff+7
        A0.q[1] = *(const uint4*)&ps[n][koff + 16];      // K = 16+koff..
        A1.q[0] = *(const uint4*)&ps[n][32 + koff];
        A1.q[1] = *(const uint4*)&ps[n][32 + koff + 16];

        // ---- stage4: acc = p @ E for our 16 columns ----
        v8f acc = {};
        acc = __builtin_amdgcn_wmma_f32_16x16x32_f16(
            false, A0.v, false, Bf0.v, (short)0, acc, false, false);
        acc = __builtin_amdgcn_wmma_f32_16x16x32_f16(
            false, A1.v, false, Bf1.v, (short)0, acc, false, false);

        // ---- prefetch em/mask for t+1 (clamped; covered by next iter) ----
        float         evn[8];
        unsigned char mkn[8];
        {
            int t2 = (t + 1 < Tv) ? (t + 1) : (Tv - 1);
            for (int r = 0; r < 8; ++r) {
                int m = r + 8 * hi;
                size_t row = (size_t)(b0 + m) * Tv + t2;
                evn[r] = em[row * Kv + jcol];
                mkn[r] = mask[row];
            }
        }

        // ---- stage5: next = rm + log(acc) + em[t]; apply mask ----
        for (int r = 0; r < 8; ++r) {
            float ns = rm[r] + fast_log(acc[r]) + evb[r];
            S[r] = mkb[r] ? ns : S[r];
        }
        for (int r = 0; r < 8; ++r) { evb[r] = evn[r]; mkb[r] = mkn[r]; }
    }

    // ---- finalize: partition[b] = logsumexp_j(S + end[j]) ----
    const float endv = endT[jcol];
    float F[8];
    for (int r = 0; r < 8; ++r) {
        F[r] = S[r] + endv;
        float v = F[r];
        v = fmaxf(v, __shfl_xor(v, 1));
        v = fmaxf(v, __shfl_xor(v, 2));
        v = fmaxf(v, __shfl_xor(v, 4));
        v = fmaxf(v, __shfl_xor(v, 8));
        if (n == 0) pmax[r + 8 * hi][wave] = v;
    }
    __syncthreads();
    for (int r = 0; r < 8; ++r) {
        int m = r + 8 * hi;
        float4 q = *(const float4*)pmax[m];
        float rmg = fmaxf(fmaxf(q.x, q.y), fmaxf(q.z, q.w));
        float s = fast_exp(F[r] - rmg);
        s += __shfl_xor(s, 1);
        s += __shfl_xor(s, 2);
        s += __shfl_xor(s, 4);
        s += __shfl_xor(s, 8);
        if (n == 0) psum[m][wave] = s;
    }
    __syncthreads();
    if (tid < 16) {
        float4 qm = *(const float4*)pmax[tid];
        float rmg = fmaxf(fmaxf(qm.x, qm.y), fmaxf(qm.z, qm.w));
        float4 qs = *(const float4*)psum[tid];
        part[b0 + tid] = rmg + fast_log(qs.x + qs.y + qs.z + qs.w);
    }
}

// ---------------------------------------------------------------------------
// Gold path score: one block per batch row, t-strided accumulation.
// ---------------------------------------------------------------------------
__global__ __launch_bounds__(256) void crf_gold_kernel(
    const float* __restrict__ em, const float* __restrict__ trans,
    const float* __restrict__ startT, const float* __restrict__ endT,
    const int* __restrict__ tags, const unsigned char* __restrict__ mask,
    float* __restrict__ gold)
{
    __shared__ float ssum[256];
    __shared__ int   scnt[256];
    const int b = blockIdx.x, tid = threadIdx.x;
    const int* tg = tags + (size_t)b * Tv;
    const unsigned char* mk = mask + (size_t)b * Tv;
    const float* eb = em + (size_t)b * Tv * Kv;

    float s = 0.f; int c = 0;
    for (int t = tid; t < Tv; t += 256) {
        int cur = tg[t];
        c += mk[t] ? 1 : 0;
        if (t >= 1 && mk[t]) {
            int prev = tg[t - 1];
            s += trans[prev * Kv + cur] + eb[(size_t)t * Kv + cur];
        }
    }
    if (tid == 0) s += startT[tg[0]] + eb[tg[0]];
    ssum[tid] = s; scnt[tid] = c;
    __syncthreads();
    for (int off = 128; off > 0; off >>= 1) {
        if (tid < off) { ssum[tid] += ssum[tid + off]; scnt[tid] += scnt[tid + off]; }
        __syncthreads();
    }
    if (tid == 0) {
        int last = tg[scnt[0] - 1];
        gold[b] = ssum[0] + endT[last];
    }
}

// ---------------------------------------------------------------------------
// Final mean over batch of (partition - gold).
// ---------------------------------------------------------------------------
__global__ __launch_bounds__(256) void crf_reduce_kernel(
    const float* __restrict__ part, const float* __restrict__ gold,
    float* __restrict__ out)
{
    __shared__ float buf[256];
    const int tid = threadIdx.x;
    float s = (part[tid] - gold[tid]) + (part[tid + 256] - gold[tid + 256]);
    buf[tid] = s;
    __syncthreads();
    for (int off = 128; off > 0; off >>= 1) {
        if (tid < off) buf[tid] += buf[tid + off];
        __syncthreads();
    }
    if (tid == 0) out[0] = buf[0] * (1.0f / (float)Bv);
}

extern "C" void kernel_launch(void* const* d_in, const int* in_sizes, int n_in,
                              void* d_out, int out_size, void* d_ws, size_t ws_size,
                              hipStream_t stream) {
    const float*         em     = (const float*)d_in[0];         // (B,T,K)
    const float*         trans  = (const float*)d_in[1];         // (K,K)
    const float*         startT = (const float*)d_in[2];         // (K,)
    const float*         endT   = (const float*)d_in[3];         // (K,)
    const int*           tags   = (const int*)d_in[4];           // (B,T)
    const unsigned char* mask   = (const unsigned char*)d_in[5]; // (B,T) bool

    float* part = (float*)d_ws;        // [512]
    float* gold = part + Bv;           // [512]
    float* out  = (float*)d_out;       // [1]

    crf_forward_kernel<<<Bv / 16, 128, 0, stream>>>(em, trans, startT, endT, mask, part);
    crf_gold_kernel<<<Bv, 256, 0, stream>>>(em, trans, startT, endT, tags, mask, gold);
    crf_reduce_kernel<<<1, 256, 0, stream>>>(part, gold, out);
}
